// Model_3272765079733
// MI455X (gfx1250) — matrix operand order repaired and checked by reading, not compiled, not run on hardware
//
#include <hip/hip_runtime.h>

// ---------------------------------------------------------------------------
// Types for gfx1250 WMMA (wave32): v_wmma_f32_16x16x32_bf16
// ---------------------------------------------------------------------------
typedef __attribute__((ext_vector_type(8)))  float  v8f;
typedef __attribute__((ext_vector_type(8)))  __bf16 v8bf;
typedef __attribute__((ext_vector_type(16))) __bf16 v16bf;

union U16 { v16bf v; v8bf h[2]; };

constexpr int BM = 128, BN = 64, BK = 32;     // block tile
constexpr int TT = 1024, NB = 16, FF = 1024;  // T, B, F
constexpr int BT = NB * TT;                   // 16384 rows

// ---------------------------------------------------------------------------
// WMMA GEMM:  C[M,N] = act( A[M,K] @ Bt[N,K]^T + bias[N] + addm[M,N] )
//   A, Bt bf16 row-major. Outputs: fp32 Cf, bf16 Cb, bf16-transposed CbT.
//   256 threads = 8 waves; each wave owns a 32x32 patch = 2x2 WMMA tiles.
//   Branch-free staging: OOB rows are CLAMPED (not zeroed) -- garbage lands
//   only in accumulator columns/rows that the guarded epilogue never stores.
//   Ping-pong LDS: one barrier per K-step, global loads overlap WMMA.
// ---------------------------------------------------------------------------
__global__ __launch_bounds__(256)
void gemm_wmma_bf16(const __bf16* __restrict__ A, int lda, long aBS,
                    const __bf16* __restrict__ Bt, int ldb, long bBS,
                    const float* __restrict__ bias,
                    const float* __restrict__ addm, int ldadd, long addBS,
                    float* __restrict__ Cf, int ldc, long cBS,
                    __bf16* __restrict__ Cb, int ldcb, long cbBS,
                    __bf16* __restrict__ CbT, int ldcbt,
                    int M, int N, int K, int relu)
{
    __shared__ __bf16 As[2][BM * BK];   // 2 x 8 KB
    __shared__ __bf16 Bs[2][BN * BK];   // 2 x 4 KB

    const int z = blockIdx.z;
    A  += (long)z * aBS;
    Bt += (long)z * bBS;
    if (addm) addm += (long)z * addBS;
    if (Cf)   Cf   += (long)z * cBS;
    if (Cb)   Cb   += (long)z * cbBS;

    const int m0  = blockIdx.y * BM;
    const int n0  = blockIdx.x * BN;
    const int tid = threadIdx.x;
    const int lane = tid & 31;
    const int wid  = tid >> 5;
    const int wm   = wid & 3;      // 4 waves along M
    const int wn   = wid >> 2;     // 2 waves along N
    const int l15  = lane & 15;
    const int hi   = lane >> 4;

    // ---- branch-free staging addresses (row indices clamped in-bounds) ----
    // A tile: each thread stages 16 contiguous halfs of one row.
    const int arow = m0 + (tid >> 1);
    const int acol = (tid & 1) * 16;
    const __bf16* aSrc = A + (long)min(arow, M - 1) * lda + acol;
    const int aOff = (tid >> 1) * BK + acol;
    // Bt tile: each thread stages 8 halfs of one row.
    const int brow = n0 + (tid >> 2);
    const int bcol = (tid & 3) * 8;
    const __bf16* bSrc = Bt + (long)min(brow, N - 1) * ldb + bcol;
    const int bOff = (tid >> 2) * BK + bcol;

    v8f acc[2][2];
#pragma unroll
    for (int i = 0; i < 2; ++i)
#pragma unroll
        for (int j = 0; j < 2; ++j)
#pragma unroll
            for (int r = 0; r < 8; ++r) acc[i][j][r] = 0.0f;

    // ---- prologue: stage first tile ----
    {
        v8bf a0 = *reinterpret_cast<const v8bf*>(aSrc);
        v8bf a1 = *reinterpret_cast<const v8bf*>(aSrc + 8);
        v8bf b0 = *reinterpret_cast<const v8bf*>(bSrc);
        *reinterpret_cast<v8bf*>(&As[0][aOff])     = a0;
        *reinterpret_cast<v8bf*>(&As[0][aOff + 8]) = a1;
        *reinterpret_cast<v8bf*>(&Bs[0][bOff])     = b0;
    }
    __syncthreads();

    int buf = 0;
    for (int k0 = 0; k0 < K; k0 += BK) {
        // ---- stage next tile into alternate buffer (overlaps WMMA) ----
        if (k0 + BK < K) {
            v8bf a0 = *reinterpret_cast<const v8bf*>(aSrc + k0 + BK);
            v8bf a1 = *reinterpret_cast<const v8bf*>(aSrc + k0 + BK + 8);
            v8bf b0 = *reinterpret_cast<const v8bf*>(bSrc + k0 + BK);
            *reinterpret_cast<v8bf*>(&As[buf ^ 1][aOff])     = a0;
            *reinterpret_cast<v8bf*>(&As[buf ^ 1][aOff + 8]) = a1;
            *reinterpret_cast<v8bf*>(&Bs[buf ^ 1][bOff])     = b0;
        }

        // ---- fragments per ISA VGPR layouts ----
        const __bf16* asb = As[buf];
        const __bf16* bsb = Bs[buf];
        v16bf af[2], bfrg[2];
#pragma unroll
        for (int sm = 0; sm < 2; ++sm) {
            int m = wm * 32 + sm * 16 + l15;
            U16 u;
            u.h[0] = *reinterpret_cast<const v8bf*>(asb + m * BK + hi * 8);       // K = hi*8 .. +7
            u.h[1] = *reinterpret_cast<const v8bf*>(asb + m * BK + 16 + hi * 8);  // K = 16+hi*8 .. +7
            af[sm] = u.v;
        }
#pragma unroll
        for (int sn = 0; sn < 2; ++sn) {
            int n = wn * 32 + sn * 16 + l15;
            U16 u;
            u.h[0] = *reinterpret_cast<const v8bf*>(bsb + n * BK + hi * 8);       // K = hi*8 .. +7
            u.h[1] = *reinterpret_cast<const v8bf*>(bsb + n * BK + 16 + hi * 8);  // K = 16+hi*8 .. +7
            bfrg[sn] = u.v;
        }
#pragma unroll
        for (int sm = 0; sm < 2; ++sm)
#pragma unroll
            for (int sn = 0; sn < 2; ++sn)
                acc[sm][sn] = __builtin_amdgcn_wmma_f32_16x16x32_bf16(
                    false, af[sm], false, bfrg[sn], (short)0, acc[sm][sn], false, false);

        __syncthreads();   // one barrier per K-step (ping-pong)
        buf ^= 1;
    }

    // ---- epilogue: bias + residual + relu, multi-format store ----
#pragma unroll
    for (int sm = 0; sm < 2; ++sm)
#pragma unroll
        for (int sn = 0; sn < 2; ++sn) {
            int nG = n0 + wn * 32 + sn * 16 + l15;
#pragma unroll
            for (int r = 0; r < 8; ++r) {
                int mG = m0 + wm * 32 + sm * 16 + hi * 8 + r;
                if (mG < M && nG < N) {
                    float v = acc[sm][sn][r];
                    if (bias) v += bias[nG];
                    if (addm) v += addm[(long)mG * ldadd + nG];
                    if (relu) v = fmaxf(v, 0.0f);
                    if (Cf)  Cf[(long)mG * ldc + nG] = v;
                    if (Cb)  Cb[(long)mG * ldcb + nG] = (__bf16)v;
                    if (CbT) CbT[(long)nG * ldcbt + mG] = (__bf16)v;
                }
            }
        }
}

// ---------------------------------------------------------------------------
// Small helper kernels
// ---------------------------------------------------------------------------
__global__ void tconv_k(const float* __restrict__ src, __bf16* __restrict__ dst,
                        int R, int C)  // src[R][C] fp32 -> dst[C][R] bf16
{
    int i = blockIdx.x * blockDim.x + threadIdx.x;
    if (i < R * C) {
        int r = i / C, c = i % C;
        dst[(long)c * R + r] = (__bf16)src[i];
    }
}

__global__ void f2bf_k(const float* __restrict__ src, __bf16* __restrict__ dst, long n)
{
    long i = (long)blockIdx.x * blockDim.x + threadIdx.x;
    if (i < n) dst[i] = (__bf16)src[i];
}

__global__ __launch_bounds__(256)
void rownorm_k(const float* __restrict__ x, float* __restrict__ xn)  // rows of F=1024
{
    int row = blockIdx.x;
    const float* p = x + (long)row * FF;
    float s = 0.0f;
    for (int j = threadIdx.x; j < FF; j += 256) { float v = p[j]; s += v * v; }
    __shared__ float red[256];
    red[threadIdx.x] = s; __syncthreads();
    for (int o = 128; o > 0; o >>= 1) {
        if (threadIdx.x < o) red[threadIdx.x] += red[threadIdx.x + o];
        __syncthreads();
    }
    if (threadIdx.x == 0) xn[row] = sqrtf(red[0]);
}

// causal conv1d(32->1, k=5, left pad 4) + sigmoid
__global__ void conv_sig_k(const float* __restrict__ a32, const float* __restrict__ cw,
                           const float* __restrict__ cb, float* __restrict__ logits,
                           float* __restrict__ s)
{
    int m = blockIdx.x * blockDim.x + threadIdx.x;
    if (m >= BT) return;
    int b = m >> 10, t = m & (TT - 1);
    float acc = cb[0];
#pragma unroll
    for (int h = 0; h < 5; ++h) {
        int tt = t + h - 4;
        if (tt >= 0) {
            const float* ap = a32 + ((long)b * TT + tt) * 32;
#pragma unroll
            for (int c = 0; c < 32; ++c) acc += ap[c] * cw[h * 32 + c];
        }
    }
    logits[m] = acc;
    s[m] = 1.0f / (1.0f + __expf(-acc));
}

__global__ void disadj_k(__bf16* __restrict__ out)
{
    long i = (long)blockIdx.x * blockDim.x + threadIdx.x;
    if (i < (long)TT * TT) {
        int r = (int)(i >> 10), c = (int)(i & (TT - 1));
        out[i] = (__bf16)__expf(-fabsf((float)(r - c)) * 0.36787944117144233f);  // 1/e
    }
}

// cosine-sim adjacency: normalize, threshold 0.7, masked softmax -> bf16 (one batch)
__global__ __launch_bounds__(256)
void adj_softmax_k(const float* __restrict__ sim, const float* __restrict__ xnb,
                   const int* __restrict__ seq_len, int b, __bf16* __restrict__ out)
{
    int t = blockIdx.x;
    int len = seq_len[b];
    bool vt = t < len;
    float xi = xnb[t];
    float vals[4]; bool msk[4];
    float mx = -3.0e38f;
#pragma unroll
    for (int k = 0; k < 4; ++k) {
        int j = threadIdx.x + k * 256;
        bool mk = vt && (j < len);
        float v = -1.0e30f;
        if (mk) {
            v = sim[(long)t * TT + j] / (xi * xnb[j] + 1e-20f);
            v = (v > 0.7f) ? v : 0.0f;
        }
        vals[k] = v; msk[k] = mk; mx = fmaxf(mx, v);
    }
    __shared__ float r1[256], r2[256];
    r1[threadIdx.x] = mx; __syncthreads();
    for (int o = 128; o > 0; o >>= 1) {
        if (threadIdx.x < o) r1[threadIdx.x] = fmaxf(r1[threadIdx.x], r1[threadIdx.x + o]);
        __syncthreads();
    }
    mx = r1[0];
    float e[4]; float sum = 0.0f;
#pragma unroll
    for (int k = 0; k < 4; ++k) { e[k] = msk[k] ? __expf(vals[k] - mx) : 0.0f; sum += e[k]; }
    r2[threadIdx.x] = sum; __syncthreads();
    for (int o = 128; o > 0; o >>= 1) {
        if (threadIdx.x < o) r2[threadIdx.x] += r2[threadIdx.x + o];
        __syncthreads();
    }
    float inv = 1.0f / fmaxf(r2[0], 1e-20f);
#pragma unroll
    for (int k = 0; k < 4; ++k) {
        int j = threadIdx.x + k * 256;
        out[(long)t * TT + j] = (__bf16)(e[k] * inv);
    }
}

// score adjacency: sco = sigmoid((1-|s_i-s_j|-0.5)/0.1), masked softmax -> bf16
__global__ __launch_bounds__(256)
void sco_softmax_k(const float* __restrict__ s, const int* __restrict__ seq_len,
                   __bf16* __restrict__ out)
{
    int t = blockIdx.x, b = blockIdx.y;
    int len = seq_len[b];
    bool vt = t < len;
    const float* sb = s + (long)b * TT;
    float st = sb[t];
    float vals[4]; bool msk[4];
    float mx = -3.0e38f;
#pragma unroll
    for (int k = 0; k < 4; ++k) {
        int j = threadIdx.x + k * 256;
        bool mk = vt && (j < len);
        float v = -1.0e30f;
        if (mk) {
            float sco = 1.0f - fabsf(st - sb[j]);
            v = 1.0f / (1.0f + __expf(-(sco - 0.5f) * 10.0f));
        }
        vals[k] = v; msk[k] = mk; mx = fmaxf(mx, v);
    }
    __shared__ float r1[256], r2[256];
    r1[threadIdx.x] = mx; __syncthreads();
    for (int o = 128; o > 0; o >>= 1) {
        if (threadIdx.x < o) r1[threadIdx.x] = fmaxf(r1[threadIdx.x], r1[threadIdx.x + o]);
        __syncthreads();
    }
    mx = r1[0];
    float e[4]; float sum = 0.0f;
#pragma unroll
    for (int k = 0; k < 4; ++k) { e[k] = msk[k] ? __expf(vals[k] - mx) : 0.0f; sum += e[k]; }
    r2[threadIdx.x] = sum; __syncthreads();
    for (int o = 128; o > 0; o >>= 1) {
        if (threadIdx.x < o) r2[threadIdx.x] += r2[threadIdx.x + o];
        __syncthreads();
    }
    float inv = 1.0f / fmaxf(r2[0], 1e-20f);
    __bf16* op = out + ((long)b * TT + t) * TT;
#pragma unroll
    for (int k = 0; k < 4; ++k) {
        int j = threadIdx.x + k * 256;
        op[j] = (__bf16)(e[k] * inv);
    }
}

// classifier: out = concat(x1,x2,x3)[.,96] @ clf_w + clf_b
__global__ void clf_k(const float* __restrict__ x1, const float* __restrict__ x2,
                      const float* __restrict__ x3, const float* __restrict__ w,
                      const float* __restrict__ bsc, float* __restrict__ out)
{
    int m = blockIdx.x * blockDim.x + threadIdx.x;
    if (m >= BT) return;
    const float* p1 = x1 + (long)m * 32;
    const float* p2 = x2 + (long)m * 32;
    const float* p3 = x3 + (long)m * 32;
    float acc = bsc[0];
#pragma unroll
    for (int c = 0; c < 32; ++c)
        acc += p1[c] * w[c] + p2[c] * w[32 + c] + p3[c] * w[64 + c];
    out[m] = acc;
}

// ---------------------------------------------------------------------------
// Host orchestration
// ---------------------------------------------------------------------------
extern "C" void kernel_launch(void* const* d_in, const int* in_sizes, int n_in,
                              void* d_out, int out_size, void* d_ws, size_t ws_size,
                              hipStream_t stream)
{
    (void)in_sizes; (void)n_in; (void)out_size; (void)ws_size;

    const float* inputs  = (const float*)d_in[0];
    const int*   seq_len = (const int*)d_in[1];
    const float* w1 = (const float*)d_in[2];   const float* b1 = (const float*)d_in[3];
    const float* w2 = (const float*)d_in[4];   const float* b2 = (const float*)d_in[5];
    const float* aw1 = (const float*)d_in[6];  const float* ab1 = (const float*)d_in[7];
    const float* aw2 = (const float*)d_in[8];  const float* ab2 = (const float*)d_in[9];
    const float* cw = (const float*)d_in[10];  const float* cb = (const float*)d_in[11];
    const float* gc1_w = (const float*)d_in[12]; const float* gc1_rw = (const float*)d_in[13];
    const float* gc1_rb = (const float*)d_in[14]; const float* gc2_w = (const float*)d_in[15];
    const float* gc3_w = (const float*)d_in[16]; const float* gc3_rw = (const float*)d_in[17];
    const float* gc3_rb = (const float*)d_in[18]; const float* gc4_w = (const float*)d_in[19];
    const float* gc5_w = (const float*)d_in[20]; const float* gc5_rw = (const float*)d_in[21];
    const float* gc5_rb = (const float*)d_in[22]; const float* gc6_w = (const float*)d_in[23];
    const float* clf_w = (const float*)d_in[24]; const float* clf_b = (const float*)d_in[25];

    float* out_f  = (float*)d_out;     // [B,T,1]
    float* logits = out_f + BT;        // [B,T,1]

    char* ws = (char*)d_ws; size_t off = 0;
    auto alloc = [&](size_t bytes) -> void* {
        void* p = ws + off; off = (off + bytes + 255) & ~(size_t)255; return p;
    };

    __bf16* in_b  = (__bf16*)alloc((size_t)BT * FF * 2);
    __bf16* w1t   = (__bf16*)alloc((size_t)512 * 1024 * 2);
    __bf16* w2t   = (__bf16*)alloc((size_t)128 * 512 * 2);
    __bf16* aw1t  = (__bf16*)alloc((size_t)64 * 128 * 2);
    __bf16* aw2t  = (__bf16*)alloc((size_t)32 * 64 * 2);
    __bf16* g1wt  = (__bf16*)alloc((size_t)32 * 128 * 2);
    __bf16* g1rwt = (__bf16*)alloc((size_t)32 * 128 * 2);
    __bf16* g2wt  = (__bf16*)alloc((size_t)32 * 32 * 2);
    __bf16* g3wt  = (__bf16*)alloc((size_t)32 * 128 * 2);
    __bf16* g3rwt = (__bf16*)alloc((size_t)32 * 128 * 2);
    __bf16* g4wt  = (__bf16*)alloc((size_t)32 * 32 * 2);
    __bf16* g5wt  = (__bf16*)alloc((size_t)32 * 128 * 2);
    __bf16* g5rwt = (__bf16*)alloc((size_t)32 * 128 * 2);
    __bf16* g6wt  = (__bf16*)alloc((size_t)32 * 32 * 2);
    __bf16* x512b = (__bf16*)alloc((size_t)BT * 512 * 2);
    float*  x128f = (float*) alloc((size_t)BT * 128 * 4);
    __bf16* x128b = (__bf16*)alloc((size_t)BT * 128 * 2);
    __bf16* a64b  = (__bf16*)alloc((size_t)BT * 64 * 2);
    float*  a32f  = (float*) alloc((size_t)BT * 32 * 4);
    float*  sbuf  = (float*) alloc((size_t)BT * 4);
    float*  xn    = (float*) alloc((size_t)BT * 4);
    float*  simbf = (float*) alloc((size_t)TT * TT * 4);
    __bf16* adjb  = (__bf16*)alloc((size_t)NB * TT * TT * 2);
    __bf16* scob  = (__bf16*)alloc((size_t)NB * TT * TT * 2);
    __bf16* disb  = (__bf16*)alloc((size_t)TT * TT * 2);
    __bf16* hwt   = (__bf16*)alloc((size_t)32 * BT * 2);
    float*  resb  = (float*) alloc((size_t)BT * 32 * 4);
    float*  x1f   = (float*) alloc((size_t)BT * 32 * 4);
    __bf16* x1b   = (__bf16*)alloc((size_t)BT * 32 * 2);
    float*  xo1   = (float*) alloc((size_t)BT * 32 * 4);
    float*  xo2   = (float*) alloc((size_t)BT * 32 * 4);
    float*  xo3   = (float*) alloc((size_t)BT * 32 * 4);

    auto gemm = [&](const __bf16* A, int lda, long aBS,
                    const __bf16* Bt, int ldb, long bBS,
                    const float* bias, const float* addm, int ldadd, long addBS,
                    float* Cf, int ldc, long cBS,
                    __bf16* Cb, int ldcb, long cbBS,
                    __bf16* CbT, int ldcbt,
                    int M, int N, int K, int relu, int batches) {
        dim3 g((N + BN - 1) / BN, (M + BM - 1) / BM, batches);
        gemm_wmma_bf16<<<g, 256, 0, stream>>>(A, lda, aBS, Bt, ldb, bBS, bias,
                                              addm, ldadd, addBS, Cf, ldc, cBS,
                                              Cb, ldcb, cbBS, CbT, ldcbt, M, N, K, relu);
    };
    auto tconv = [&](const float* src, __bf16* dst, int R, int C) {
        int n = R * C;
        tconv_k<<<(n + 255) / 256, 256, 0, stream>>>(src, dst, R, C);
    };

    // --- weight transposes (fp32 [K,N] -> bf16 [N,K]) ---
    tconv(w1, w1t, 1024, 512);   tconv(w2, w2t, 512, 128);
    tconv(aw1, aw1t, 128, 64);   tconv(aw2, aw2t, 64, 32);
    tconv(gc1_w, g1wt, 128, 32); tconv(gc1_rw, g1rwt, 128, 32); tconv(gc2_w, g2wt, 32, 32);
    tconv(gc3_w, g3wt, 128, 32); tconv(gc3_rw, g3rwt, 128, 32); tconv(gc4_w, g4wt, 32, 32);
    tconv(gc5_w, g5wt, 128, 32); tconv(gc5_rw, g5rwt, 128, 32); tconv(gc6_w, g6wt, 32, 32);

    // --- inputs -> bf16; row norms ---
    {
        long n = (long)BT * FF;
        f2bf_k<<<(unsigned)((n + 255) / 256), 256, 0, stream>>>(inputs, in_b, n);
    }
    rownorm_k<<<BT, 256, 0, stream>>>(inputs, xn);

    // --- stem (all WMMA) ---
    gemm(in_b, FF, 0,  w1t, 1024, 0, b1, nullptr, 0, 0,
         nullptr, 0, 0, x512b, 512, 0, nullptr, 0, BT, 512, 1024, 1, 1);
    gemm(x512b, 512, 0, w2t, 512, 0, b2, nullptr, 0, 0,
         x128f, 128, 0, x128b, 128, 0, nullptr, 0, BT, 128, 512, 1, 1);
    gemm(x128b, 128, 0, aw1t, 128, 0, ab1, nullptr, 0, 0,
         nullptr, 0, 0, a64b, 64, 0, nullptr, 0, BT, 64, 128, 1, 1);
    gemm(a64b, 64, 0,  aw2t, 64, 0, ab2, nullptr, 0, 0,
         a32f, 32, 0, nullptr, 0, 0, nullptr, 0, BT, 32, 64, 1, 1);

    // --- causal conv + sigmoid (writes logits output directly) ---
    conv_sig_k<<<(BT + 255) / 256, 256, 0, stream>>>(a32f, cw, cb, logits, sbuf);

    // --- adjacencies ---
    disadj_k<<<((TT * TT) + 255) / 256, 256, 0, stream>>>(disb);
    sco_softmax_k<<<dim3(TT, NB), 256, 0, stream>>>(sbuf, seq_len, scob);
    for (int b = 0; b < NB; ++b) {   // sim = inputs @ inputs^T, per batch (WMMA), fused softmax
        const __bf16* ib = in_b + (size_t)b * TT * FF;
        gemm(ib, FF, 0, ib, FF, 0, nullptr, nullptr, 0, 0,
             simbf, TT, 0, nullptr, 0, 0, nullptr, 0, TT, TT, FF, 0, 1);
        adj_softmax_k<<<TT, 256, 0, stream>>>(simbf, xn + b * TT, seq_len, b,
                                              adjb + (size_t)b * TT * TT);
    }

    // --- graph-conv branch: two GC layers with residuals (all WMMA) ---
    auto branch = [&](const __bf16* adjA, long adjStride, const __bf16* wT,
                      const __bf16* rwT, const float* rb, const __bf16* w2T, float* xo) {
        // res = h @ rw + rb
        gemm(x128b, 128, 0, rwT, 128, 0, rb, nullptr, 0, 0,
             resb, 32, 0, nullptr, 0, 0, nullptr, 0, BT, 32, 128, 0, 1);
        // hwt = (h @ W)^T  (bf16 transposed out, [32][B*T])
        gemm(x128b, 128, 0, wT, 128, 0, nullptr, nullptr, 0, 0,
             nullptr, 0, 0, nullptr, 0, 0, hwt, BT, BT, 32, 128, 0, 1);
        // x1 = relu(A @ hW + res)   per batch
        gemm(adjA, TT, adjStride, hwt, BT, TT, nullptr, resb, 32, (long)TT * 32,
             x1f, 32, (long)TT * 32, x1b, 32, (long)TT * 32, nullptr, 0,
             TT, 32, TT, 1, NB);
        // hwt = (x1 @ W2)^T
        gemm(x1b, 32, 0, w2T, 32, 0, nullptr, nullptr, 0, 0,
             nullptr, 0, 0, nullptr, 0, 0, hwt, BT, BT, 32, 32, 0, 1);
        // xo = relu(A @ x1W2 + x1)  per batch
        gemm(adjA, TT, adjStride, hwt, BT, TT, nullptr, x1f, 32, (long)TT * 32,
             xo, 32, (long)TT * 32, nullptr, 0, 0, nullptr, 0,
             TT, 32, TT, 1, NB);
    };
    branch(adjb, (long)TT * TT, g1wt, g1rwt, gc1_rb, g2wt, xo1);
    branch(disb, 0,             g3wt, g3rwt, gc3_rb, g4wt, xo2);
    branch(scob, (long)TT * TT, g5wt, g5rwt, gc5_rb, g6wt, xo3);

    // --- classifier ---
    clf_k<<<(BT + 255) / 256, 256, 0, stream>>>(xo1, xo2, xo3, clf_w, clf_b, out_f);
}
